// MultiLayerCrossAttention_43215960932904
// MI455X (gfx1250) — matrix operand order repaired and checked by reading, not compiled, not run on hardware
//
#include <hip/hip_runtime.h>
#include <cstdint>

// ---------------------------------------------------------------------------
// MultiLayerCrossAttention for MI455X (gfx1250, wave32)
//   - all matmuls on v_wmma_f32_16x16x32_f16
//   - double-buffered LDS tiles fed by GLOBAL_LOAD_ASYNC_TO_LDS_B128 (ASYNCcnt)
//   - DPP row_xmask butterflies for softmax row reductions (no DS traffic)
// ---------------------------------------------------------------------------

typedef __attribute__((ext_vector_type(16))) _Float16 v16h;
typedef __attribute__((ext_vector_type(8)))  _Float16 v8h;
typedef __attribute__((ext_vector_type(8)))  float    v8f;

constexpr int cB  = 4;
constexpr int cT  = 288;
constexpr int cN  = 32;
constexpr int cD  = 512;
constexpr int cF  = 2048;
constexpr int cH  = 8;
constexpr int cHD = 64;
constexpr int cL  = 3;
constexpr int M_ROWS = cB * cN * cT;   // 36864

union F16Frag { v16h v; v8h h[2]; };

// ---------------- CDNA5 async global->LDS (guarded, sync fallback) ----------
#if __has_builtin(__builtin_amdgcn_global_load_async_to_lds_b128)
#define HAVE_ASYNC 1
#else
#define HAVE_ASYNC 0
#endif

typedef int gvec4i __attribute__((vector_size(16)));
typedef __attribute__((address_space(1))) gvec4i as1_v4i;
typedef __attribute__((address_space(3))) gvec4i as3_v4i;

__device__ __forceinline__ void copy16(void* lds, const void* g) {
#if HAVE_ASYNC
  __builtin_amdgcn_global_load_async_to_lds_b128(
      (as1_v4i*)(uintptr_t)g,
      (as3_v4i*)(uint32_t)(uintptr_t)lds,
      0, 0);
#else
  *reinterpret_cast<v8h*>(lds) = *reinterpret_cast<const v8h*>(g);
#endif
}

__device__ __forceinline__ void wait_async0() {
#if HAVE_ASYNC
#if __has_builtin(__builtin_amdgcn_s_wait_asynccnt)
  __builtin_amdgcn_s_wait_asynccnt(0);
#else
  asm volatile("s_wait_asynccnt 0" ::: "memory");
#endif
#endif
}

__device__ __forceinline__ void wait_lds() {
#if __has_builtin(__builtin_amdgcn_s_wait_dscnt)
  __builtin_amdgcn_s_wait_dscnt(0);
#else
  asm volatile("s_wait_dscnt 0" ::: "memory");
#endif
}

// ---------------- DPP row_xmask reductions over 16-lane groups --------------
template <int CTRL>
__device__ __forceinline__ float dpp_mov(float x) {
  return __builtin_bit_cast(float, __builtin_amdgcn_update_dpp(
      0, __builtin_bit_cast(int, x), CTRL, 0xf, 0xf, true));
}
__device__ __forceinline__ float red16_max(float x) {   // ROW_XMASK = 0x160|m
  x = fmaxf(x, dpp_mov<0x161>(x));
  x = fmaxf(x, dpp_mov<0x162>(x));
  x = fmaxf(x, dpp_mov<0x164>(x));
  x = fmaxf(x, dpp_mov<0x168>(x));
  return x;
}
__device__ __forceinline__ float red16_sum(float x) {
  x += dpp_mov<0x161>(x);
  x += dpp_mov<0x162>(x);
  x += dpp_mov<0x164>(x);
  x += dpp_mov<0x168>(x);
  return x;
}

// --------------------------- fp32 -> fp16 convert ---------------------------
__global__ __launch_bounds__(256) void cvt_kernel(const float* __restrict__ src,
                                                  _Float16* __restrict__ dst, int n) {
  int i = blockIdx.x * 256 + threadIdx.x;
  if (i < n) dst[i] = (_Float16)src[i];
}

// ------------------ fp32 -> fp16 convert + transpose (K,N)->(N,K) -----------
__global__ __launch_bounds__(256) void cvt_t_kernel(const float* __restrict__ src,
                                                    _Float16* __restrict__ dst,
                                                    int K, int N) {
  __shared__ float tile[32][33];
  const float* s = src + (size_t)blockIdx.z * K * N;
  _Float16*    d = dst + (size_t)blockIdx.z * K * N;
  int n0 = blockIdx.x * 32, k0 = blockIdx.y * 32;
  int tx = threadIdx.x & 31, ty = threadIdx.x >> 5;   // 32 x 8
#pragma unroll
  for (int i = 0; i < 32; i += 8)
    tile[ty + i][tx] = s[(size_t)(k0 + ty + i) * N + n0 + tx];
  __syncthreads();
#pragma unroll
  for (int i = 0; i < 32; i += 8)
    d[(size_t)(n0 + ty + i) * K + k0 + tx] = (_Float16)tile[tx][ty + i];
}

// ------------------- input projection (rank-1) + transpose -------------------
__global__ __launch_bounds__(256) void proj_kernel(const float* __restrict__ x,
                                                   const float* __restrict__ w,
                                                   const float* __restrict__ bias,
                                                   float* __restrict__ out32,
                                                   _Float16* __restrict__ out16) {
  int rid = blockIdx.x;             // (b*N + n)*T + t
  int t  = rid % cT;
  int bn = rid / cT;
  int n  = bn % cN;
  int b  = bn / cN;
  float xv = x[(b * cT + t) * cN + n];
  size_t base = (size_t)rid * cD;
  for (int d = threadIdx.x; d < cD; d += 256) {
    float v = xv * w[d] + bias[d];
    if (out32) out32[base + d] = v;
    out16[base + d] = (_Float16)v;
  }
}

// --------------------------------- GEMM -------------------------------------
// C = A(M x K, f16) @ W(K x N) + bias, W given pre-transposed Wt(N x K).
// Block tile 128x128, 8 waves, each a 64x32 sub-tile; double-buffered async LDS.
// OUT16T writes the f16 result transposed per (bn, head): [bn][h][e][t].
template <bool RELU, bool OUT16, bool OUT32, bool OUT16T>
__global__ __launch_bounds__(256) void gemm_kernel(const _Float16* __restrict__ A,
                                                   const _Float16* __restrict__ Wt,
                                                   const float* __restrict__ bias,
                                                   _Float16* __restrict__ C16,
                                                   float* __restrict__ C32,
                                                   int Ndim, int Kdim, float oscale) {
  __shared__ _Float16 As[2][128][40];
  __shared__ _Float16 Bt[2][128][40];

  const int tid  = threadIdx.x;
  const int lane = tid & 31;
  const int wave = tid >> 5;
  const int wm   = wave >> 2;        // 0..1
  const int wn   = wave & 3;         // 0..3
  const int m0   = blockIdx.y * 128;
  const int n0   = blockIdx.x * 128;

  const int lr  = lane & 15;
  const int lh  = lane >> 4;
  const int kbA = lh * 8;
  const int kbB = lh * 8;

  v8f acc[4][2] = {};

  // stage one 128x32 tile pair into buffer `buf` (2 A + 2 B chunks per thread)
  auto stage = [&](int buf, int k0) {
#pragma unroll
    for (int u = 0; u < 2; u++) {
      int c   = tid * 2 + u;         // 0..511
      int row = c >> 2;
      int seg = (c & 3) * 8;
      copy16(&As[buf][row][seg], &A [(size_t)(m0 + row) * Kdim + k0 + seg]);
      copy16(&Bt[buf][row][seg], &Wt[(size_t)(n0 + row) * Kdim + k0 + seg]);
    }
  };

  const int nk = Kdim / 32;
  stage(0, 0);
  for (int j = 0; j < nk; j++) {
    wait_async0();
    __syncthreads();
    if (j + 1 < nk) stage((j + 1) & 1, (j + 1) * 32);
    const int b = j & 1;

    F16Frag af[4], bf[2];
#pragma unroll
    for (int i = 0; i < 4; i++) {
      int row = wm * 64 + i * 16 + lr;
      af[i].h[0] = *reinterpret_cast<const v8h*>(&As[b][row][kbA]);
      af[i].h[1] = *reinterpret_cast<const v8h*>(&As[b][row][kbA + 16]);
    }
#pragma unroll
    for (int jj = 0; jj < 2; jj++) {
      int col = wn * 32 + jj * 16 + lr;
      bf[jj].h[0] = *reinterpret_cast<const v8h*>(&Bt[b][col][kbB]);
      bf[jj].h[1] = *reinterpret_cast<const v8h*>(&Bt[b][col][kbB + 16]);
    }
#pragma unroll
    for (int i = 0; i < 4; i++)
#pragma unroll
      for (int jj = 0; jj < 2; jj++)
        acc[i][jj] = __builtin_amdgcn_wmma_f32_16x16x32_f16(
            false, af[i].v, false, bf[jj].v, (short)0, acc[i][jj], false, false);
  }

  // epilogue
#pragma unroll
  for (int i = 0; i < 4; i++) {
#pragma unroll
    for (int jj = 0; jj < 2; jj++) {
      int n = n0 + wn * 32 + jj * 16 + lr;
      float bv = bias ? bias[n] : 0.0f;
#pragma unroll
      for (int v = 0; v < 8; v++) {
        int m = m0 + wm * 64 + i * 16 + v + lh * 8;
        float val = (acc[i][jj][v] + bv) * oscale;
        if (RELU) val = val > 0.0f ? val : 0.0f;
        if (OUT32) C32[(size_t)m * Ndim + n] = val;
        if (OUT16) C16[(size_t)m * Ndim + n] = (_Float16)val;
        if (OUT16T) {            // [bn][h][e][t] for attention V
          int bn = m / cT, row = m % cT;
          int hh = n >> 6, e = n & 63;
          C16[(((size_t)bn * cH + hh) * cHD + e) * cT + row] = (_Float16)val;
        }
      }
    }
  }
}

// ------------------------------- attention -----------------------------------
// One block: one (b,n,h), 6 waves * 16-query blocks. 32-key steps, double-
// buffered async staging; V arrives pre-transposed [bn][h][e][t].
__global__ __launch_bounds__(192) void attn_kernel(const _Float16* __restrict__ Q,
                                                   const _Float16* __restrict__ Km,
                                                   const _Float16* __restrict__ Vg,
                                                   _Float16* __restrict__ O) {
  __shared__ _Float16 Ks[2][32][72];      // [key][hd]
  __shared__ _Float16 Vt[2][64][40];      // [hd][key]
  __shared__ _Float16 Ps[6][16][40];      // per-wave P tile (16x32)

  const int tid  = threadIdx.x;
  const int lane = tid & 31;
  const int wave = tid >> 5;              // 0..5
  const int qgrp = blockIdx.x % 3;
  const int bnh  = blockIdx.x / 3;
  const int h    = bnh % cH;
  const int bn   = bnh / cH;
  const size_t base  = (size_t)bn * cT * cD + (size_t)h * cHD;
  const size_t vbase = ((size_t)bn * cH + h) * cHD * cT;

  const int q0  = (qgrp * 6 + wave) * 16;
  const int lr  = lane & 15;
  const int lh  = lane >> 4;
  const int kbA = lh * 8;
  const int kbB = lh * 8;

  // Q fragments (scale 1/sqrt(HD) already folded in by the Q projection)
  F16Frag qf[2];
  {
    const _Float16* qrow = &Q[base + (size_t)(q0 + lr) * cD];
    qf[0].h[0] = *reinterpret_cast<const v8h*>(qrow + kbA);
    qf[0].h[1] = *reinterpret_cast<const v8h*>(qrow + kbA + 16);
    qf[1].h[0] = *reinterpret_cast<const v8h*>(qrow + 32 + kbA);
    qf[1].h[1] = *reinterpret_cast<const v8h*>(qrow + 32 + kbA + 16);
  }

  v8f oacc[4] = {};
  float mi[8], li[8];
#pragma unroll
  for (int v = 0; v < 8; v++) { mi[v] = -1e30f; li[v] = 0.0f; }

  auto stage = [&](int buf, int kk) {
    if (tid < 128) {                       // K tile: 32 keys x 64, as-is
#pragma unroll
      for (int u = 0; u < 2; u++) {
        int c = tid * 2 + u;               // 0..255
        int key = c >> 3, seg = (c & 7) * 8;
        copy16(&Ks[buf][key][seg], &Km[base + (size_t)(kk + key) * cD + seg]);
      }
    }
    if (tid >= 64) {                       // V tile: 64 dims x 32 keys (pre-T)
      int t2 = tid - 64;                   // 0..127
#pragma unroll
      for (int u = 0; u < 2; u++) {
        int c = t2 * 2 + u;                // 0..255
        int e = c >> 2, seg = (c & 3) * 8;
        copy16(&Vt[buf][e][seg], &Vg[vbase + (size_t)e * cT + kk + seg]);
      }
    }
  };

  stage(0, 0);
  const int nk = cT / 32;                  // 9
  for (int j = 0; j < nk; j++) {
    wait_async0();
    __syncthreads();
    if (j + 1 < nk) stage((j + 1) & 1, (j + 1) * 32);
    const int b = j & 1;

    // S = Q K^T for 32 keys
    v8f s0 = {}, s1 = {};
#pragma unroll
    for (int c = 0; c < 2; c++) {
      F16Frag kf0, kf1;
      kf0.h[0] = *reinterpret_cast<const v8h*>(&Ks[b][lr][c * 32 + kbB]);
      kf0.h[1] = *reinterpret_cast<const v8h*>(&Ks[b][lr][c * 32 + kbB + 16]);
      kf1.h[0] = *reinterpret_cast<const v8h*>(&Ks[b][16 + lr][c * 32 + kbB]);
      kf1.h[1] = *reinterpret_cast<const v8h*>(&Ks[b][16 + lr][c * 32 + kbB + 16]);
      s0 = __builtin_amdgcn_wmma_f32_16x16x32_f16(false, qf[c].v, false, kf0.v, (short)0, s0, false, false);
      s1 = __builtin_amdgcn_wmma_f32_16x16x32_f16(false, qf[c].v, false, kf1.v, (short)0, s1, false, false);
    }

    // online softmax; row stats reduced across the 16-lane group via DPP
#pragma unroll
    for (int v = 0; v < 8; v++) {
      float a  = s0[v];
      float b2 = s1[v];
      float mx = red16_max(fmaxf(a, b2));
      float mnew  = fmaxf(mi[v], mx);
      float alpha = __expf(mi[v] - mnew);
      float p0 = __expf(a - mnew);
      float p1 = __expf(b2 - mnew);
      float rs = red16_sum(p0 + p1);
      li[v] = li[v] * alpha + rs;
      mi[v] = mnew;
#pragma unroll
      for (int e = 0; e < 4; e++) oacc[e][v] *= alpha;
      int prow = v + lh * 8;
      Ps[wave][prow][lr]      = (_Float16)p0;
      Ps[wave][prow][16 + lr] = (_Float16)p1;
    }
    wait_lds();                            // wave-local P store -> load

    F16Frag pf;
    pf.h[0] = *reinterpret_cast<const v8h*>(&Ps[wave][lr][kbA]);
    pf.h[1] = *reinterpret_cast<const v8h*>(&Ps[wave][lr][kbA + 16]);
#pragma unroll
    for (int e = 0; e < 4; e++) {
      F16Frag vf;
      vf.h[0] = *reinterpret_cast<const v8h*>(&Vt[b][e * 16 + lr][kbB]);
      vf.h[1] = *reinterpret_cast<const v8h*>(&Vt[b][e * 16 + lr][kbB + 16]);
      oacc[e] = __builtin_amdgcn_wmma_f32_16x16x32_f16(false, pf.v, false, vf.v, (short)0, oacc[e], false, false);
    }
  }

#pragma unroll
  for (int v = 0; v < 8; v++) {
    float inv = 1.0f / li[v];
    int row = q0 + v + lh * 8;
#pragma unroll
    for (int e = 0; e < 4; e++)
      O[base + (size_t)row * cD + e * 16 + lr] = (_Float16)(oacc[e][v] * inv);
  }
}

// ------------------------- fused residual + LayerNorm ------------------------
__global__ __launch_bounds__(256) void ln_kernel(const float* __restrict__ hin,
                                                 const float* __restrict__ delta,
                                                 const float* __restrict__ g,
                                                 const float* __restrict__ b,
                                                 float* __restrict__ out32,
                                                 _Float16* __restrict__ out16) {
  const int lane = threadIdx.x & 31;
  const int wave = threadIdx.x >> 5;
  const int row  = blockIdx.x * 8 + wave;
  const size_t base = (size_t)row * cD;

  float vals[16];
  float s = 0.0f;
#pragma unroll
  for (int i = 0; i < 16; i++) {
    int d = i * 32 + lane;
    float v = hin[base + d] + delta[base + d];
    vals[i] = v;
    s += v;
  }
#pragma unroll
  for (int d2 = 1; d2 < 32; d2 <<= 1) s += __shfl_xor(s, d2, 32);
  float mu = s * (1.0f / cD);
  float var = 0.0f;
#pragma unroll
  for (int i = 0; i < 16; i++) { float t = vals[i] - mu; var += t * t; }
#pragma unroll
  for (int d2 = 1; d2 < 32; d2 <<= 1) var += __shfl_xor(var, d2, 32);
  var *= (1.0f / cD);
  float rinv = rsqrtf(var + 1e-5f);
#pragma unroll
  for (int i = 0; i < 16; i++) {
    int d = i * 32 + lane;
    float o = (vals[i] - mu) * rinv * g[d] + b[d];
    out32[base + d] = o;
    out16[base + d] = (_Float16)o;
  }
}

// ------------------------------- host driver ---------------------------------
extern "C" void kernel_launch(void* const* d_in, const int* in_sizes, int n_in,
                              void* d_out, int out_size, void* d_ws, size_t ws_size,
                              hipStream_t stream) {
  const float* x   = (const float*)d_in[0];
  const float* xc  = (const float*)d_in[1];
  const float* ipw = (const float*)d_in[2];
  const float* ipb = (const float*)d_in[3];
  const float* cpw = (const float*)d_in[4];
  const float* cpb = (const float*)d_in[5];
  const float* Wq  = (const float*)d_in[6];
  const float* bq  = (const float*)d_in[7];
  const float* Wk  = (const float*)d_in[8];
  const float* bk  = (const float*)d_in[9];
  const float* Wv  = (const float*)d_in[10];
  const float* bv  = (const float*)d_in[11];
  const float* Wo  = (const float*)d_in[12];
  const float* bo  = (const float*)d_in[13];
  const float* W1  = (const float*)d_in[14];
  const float* b1  = (const float*)d_in[15];
  const float* W2  = (const float*)d_in[16];
  const float* b2  = (const float*)d_in[17];
  const float* g1  = (const float*)d_in[18];
  const float* be1 = (const float*)d_in[19];
  const float* g2  = (const float*)d_in[20];
  const float* be2 = (const float*)d_in[21];

  char* ws = (char*)d_ws;
  size_t off = 0;
  auto carve = [&](size_t bytes) -> void* {
    void* p = ws + off;
    off += (bytes + 255) & ~(size_t)255;
    return p;
  };

  float*    h32   = (float*)   carve((size_t)M_ROWS * cD * 4);
  float*    tmp32 = (float*)   carve((size_t)M_ROWS * cD * 4);
  _Float16* h16   = (_Float16*)carve((size_t)M_ROWS * cD * 2);
  _Float16* ctx16 = (_Float16*)carve((size_t)M_ROWS * cD * 2);
  _Float16* q16   = (_Float16*)carve((size_t)M_ROWS * cD * 2);
  _Float16* k16   = (_Float16*)carve((size_t)M_ROWS * cD * 2);
  _Float16* v16t  = (_Float16*)carve((size_t)M_ROWS * cD * 2);  // [bn][h][e][t]
  _Float16* o16   = (_Float16*)carve((size_t)M_ROWS * cD * 2);
  _Float16* mid16 = (_Float16*)carve((size_t)M_ROWS * cF * 2);
  const size_t wDD = (size_t)cL * cD * cD;
  const size_t wDF = (size_t)cL * cD * cF;
  _Float16* wq16 = (_Float16*)carve(wDD * 2);   // all stored transposed (N,K)
  _Float16* wk16 = (_Float16*)carve(wDD * 2);
  _Float16* wv16 = (_Float16*)carve(wDD * 2);
  _Float16* wo16 = (_Float16*)carve(wDD * 2);
  _Float16* w116 = (_Float16*)carve(wDF * 2);
  _Float16* w216 = (_Float16*)carve(wDF * 2);

  // weight conversion fp32 -> fp16, transposed (K,N)->(N,K), batched over L
  auto cvt_t = [&](const float* s, _Float16* d, int K, int N) {
    cvt_t_kernel<<<dim3(N / 32, K / 32, cL), 256, 0, stream>>>(s, d, K, N);
  };
  cvt_t(Wq, wq16, cD, cD);
  cvt_t(Wk, wk16, cD, cD);
  cvt_t(Wv, wv16, cD, cD);
  cvt_t(Wo, wo16, cD, cD);
  cvt_t(W1, w116, cD, cF);
  cvt_t(W2, w216, cF, cD);

  // input / contextual projections
  proj_kernel<<<M_ROWS, 256, 0, stream>>>(x,  ipw, ipb, h32,    h16);
  proj_kernel<<<M_ROWS, 256, 0, stream>>>(xc, cpw, cpb, nullptr, ctx16);

  const dim3 g512(cD / 128, M_ROWS / 128);   // (4, 288)
  const dim3 gF  (cF / 128, M_ROWS / 128);   // (16, 288)
  const float qscale = 0.125f;               // 1/sqrt(HD)

  for (int l = 0; l < cL; l++) {
    const _Float16* wq_l = wq16 + (size_t)l * cD * cD;
    const _Float16* wk_l = wk16 + (size_t)l * cD * cD;
    const _Float16* wv_l = wv16 + (size_t)l * cD * cD;
    const _Float16* wo_l = wo16 + (size_t)l * cD * cD;
    const _Float16* w1_l = w116 + (size_t)l * cD * cF;
    const _Float16* w2_l = w216 + (size_t)l * cF * cD;

    gemm_kernel<false, true, false, false><<<g512, 256, 0, stream>>>(
        h16,   wq_l, bq + l * cD, q16, nullptr, cD, cD, qscale);
    gemm_kernel<false, true, false, false><<<g512, 256, 0, stream>>>(
        ctx16, wk_l, bk + l * cD, k16, nullptr, cD, cD, 1.0f);
    gemm_kernel<false, false, false, true><<<g512, 256, 0, stream>>>(
        ctx16, wv_l, bv + l * cD, v16t, nullptr, cD, cD, 1.0f);

    attn_kernel<<<cB * cN * cH * 3, 192, 0, stream>>>(q16, k16, v16t, o16);

    gemm_kernel<false, false, true, false><<<g512, 256, 0, stream>>>(
        o16, wo_l, bo + l * cD, nullptr, tmp32, cD, cD, 1.0f);
    ln_kernel<<<M_ROWS / 8, 256, 0, stream>>>(h32, tmp32, g1 + l * cD, be1 + l * cD, h32, h16);

    gemm_kernel<true, true, false, false><<<gF, 256, 0, stream>>>(
        h16,   w1_l, b1 + l * cF, mid16, nullptr, cF, cD, 1.0f);
    gemm_kernel<false, false, true, false><<<g512, 256, 0, stream>>>(
        mid16, w2_l, b2 + l * cD, nullptr, tmp32, cD, cF, 1.0f);

    float* dst = (l == cL - 1) ? (float*)d_out : h32;
    ln_kernel<<<M_ROWS / 8, 256, 0, stream>>>(h32, tmp32, g2 + l * cD, be2 + l * cD, dst, h16);
  }
}
